// ODConv_28535762714881
// MI455X (gfx1250) — matrix operand order repaired and checked by reading, not compiled, not run on hardware
//
#include <hip/hip_runtime.h>
#include <hip/hip_bf16.h>
#include <stdint.h>

// ---- problem constants (from reference) ----
#define B_    16
#define CIN   256
#define COUT  256
#define KN    4
#define HID   16
#define HW    64
#define NPIX  4096      // HW*HW
#define PADW  66        // HW + 2 (pad=1 each side)
#define EPSV  1e-5f

// ---- conv tiling ----
#define MT 128          // block M tile (cout)
#define NT 128          // block N tile (pixels)
#define KT 32           // WMMA K step (bf16)
#define NSTEP 72        // 9 (r,s) x 8 cin-steps

typedef __attribute__((ext_vector_type(16))) __bf16 v16bf;
typedef __attribute__((ext_vector_type(8)))  float  v8f;

__device__ __forceinline__ unsigned short f2bf(float f) {
  unsigned int u = __float_as_uint(f);
  u += 0x7FFFu + ((u >> 16) & 1u);          // round-to-nearest-even
  return (unsigned short)(u >> 16);
}

// CDNA5 async cache->LDS copy (ASYNCcnt-tracked), bypasses VGPR staging.
__device__ __forceinline__ void async_b128_to_lds(unsigned lds_off, const void* g) {
  asm volatile("global_load_async_to_lds_b128 %0, %1, off"
               :: "v"(lds_off), "v"((unsigned long long)(uintptr_t)g)
               : "memory");
}
__device__ __forceinline__ void wait_async0() {
  asm volatile("s_wait_asynccnt 0x0" ::: "memory");
}

// ---------------------------------------------------------------------------
// Kernel 1: global average pool  x[b,c,:,:] -> pooled[b,c]
// ---------------------------------------------------------------------------
__global__ void pool_kernel(const float* __restrict__ x, float* __restrict__ pooled) {
  const int bc = blockIdx.x;                 // 0..B*CIN-1
  const float* p = x + (size_t)bc * NPIX;
  float s = 0.f;
  for (int i = threadIdx.x; i < NPIX; i += 256) s += p[i];
  __shared__ float red[256];
  red[threadIdx.x] = s;
  __syncthreads();
  for (int off = 128; off > 0; off >>= 1) {
    if (threadIdx.x < off) red[threadIdx.x] += red[threadIdx.x + off];
    __syncthreads();
  }
  if (threadIdx.x == 0) pooled[bc] = red[0] * (1.0f / NPIX);
}

// ---------------------------------------------------------------------------
// Kernel 2: attention MLP (dense 256->16, BN, ReLU, 16->4, softmax). One block.
// ---------------------------------------------------------------------------
__global__ void att_kernel(const float* __restrict__ pooled,
                           const float* __restrict__ w1,
                           const float* __restrict__ gamma,
                           const float* __restrict__ beta,
                           const float* __restrict__ mean,
                           const float* __restrict__ var,
                           const float* __restrict__ w2,
                           float* __restrict__ att) {
  __shared__ float hsm[B_ * HID];
  const int t = threadIdx.x;                 // 256 threads: (b, hid)
  const int bb = t >> 4, hh = t & 15;
  float acc = 0.f;
  const float* pr = pooled + bb * CIN;
  const float* wr = w1 + hh * CIN;
  for (int c = 0; c < CIN; ++c) acc += pr[c] * wr[c];
  acc = (acc - mean[hh]) * rsqrtf(var[hh] + EPSV) * gamma[hh] + beta[hh];
  hsm[t] = fmaxf(acc, 0.f);
  __syncthreads();
  if (t < B_) {
    float lg[KN];
    float mx = -1e30f;
    for (int k = 0; k < KN; ++k) {
      float a = 0.f;
      for (int h = 0; h < HID; ++h) a += hsm[t * HID + h] * w2[k * HID + h];
      lg[k] = a; mx = fmaxf(mx, a);
    }
    float se = 0.f;
    for (int k = 0; k < KN; ++k) { lg[k] = __expf(lg[k] - mx); se += lg[k]; }
    const float inv = 1.f / se;
    for (int k = 0; k < KN; ++k) att[t * KN + k] = lg[k] * inv;
  }
}

// ---------------------------------------------------------------------------
// Kernel 3: pad + NCHW->NHWC + f32->bf16:  xpad[b][yp][xp][c]  (halo pre-zeroed)
// ---------------------------------------------------------------------------
__global__ void xpad_kernel(const float* __restrict__ x,
                            unsigned short* __restrict__ xpad) {
  const int xc = threadIdx.x;                // 0..63 (column)
  const int y = blockIdx.x, c = blockIdx.y, b = blockIdx.z;
  const float v = x[(((size_t)(b * CIN + c)) * HW + y) * HW + xc];
  xpad[(((size_t)(b * PADW + (y + 1))) * PADW + (xc + 1)) * CIN + c] = f2bf(v);
}

// ---------------------------------------------------------------------------
// Kernel 4: per-sample weight aggregation -> bf16, layout [b][o][rs][cin]
// weight: [KN][COUT][CIN][3][3] f32
// ---------------------------------------------------------------------------
__global__ void aggw_kernel(const float* __restrict__ weight,
                            const float* __restrict__ att,
                            unsigned short* __restrict__ aggw) {
  const int i = threadIdx.x;                 // cin
  const int o = blockIdx.x, b = blockIdx.y;
  const float a0 = att[b * KN + 0], a1 = att[b * KN + 1];
  const float a2 = att[b * KN + 2], a3 = att[b * KN + 3];
  const size_t kstride = (size_t)COUT * CIN * 9;
  const float* w0 = weight + (((size_t)o) * CIN + i) * 9;
  #pragma unroll
  for (int rs = 0; rs < 9; ++rs) {
    const float v = a0 * w0[rs] + a1 * w0[kstride + rs]
                  + a2 * w0[2 * kstride + rs] + a3 * w0[3 * kstride + rs];
    aggw[(((size_t)(b * COUT + o)) * 9 + rs) * CIN + i] = f2bf(v);
  }
}

// ---------------------------------------------------------------------------
// Kernel 5: implicit-GEMM 3x3 conv via v_wmma_f32_16x16x32_bf16
//   double-buffered LDS fed by global_load_async_to_lds_b128 (ASYNCcnt),
//   one barrier per K-step. Block 256 thr = 8 waves (4M x 2N), tile 128x128x32.
// ---------------------------------------------------------------------------
__global__ __launch_bounds__(256) void conv_kernel(
    const unsigned short* __restrict__ xpad,   // [b][66][66][256] bf16
    const unsigned short* __restrict__ aggw,   // [b][256][9][256] bf16
    float* __restrict__ out) {                 // [b][256][64][64] f32
  __shared__ __align__(16) unsigned short Asm[2][MT * KT];  // [m][k] row-major
  __shared__ __align__(16) unsigned short Bsm[2][NT * KT];  // [n][k] row-major

  const int tid  = threadIdx.x;
  const int lane = tid & 31;
  const int wave = tid >> 5;
  const int wm = wave & 3;                   // wave M position (0..3)
  const int wn = wave >> 2;                  // wave N position (0..1)
  const int bx = blockIdx.x;                 // N tile (0..31)
  const int by = blockIdx.y;                 // M tile (0..1)
  const int b  = blockIdx.z;

  // ---- per-thread staging bases (2x 16B chunks each for A and B) ----
  const unsigned short* agbase[2];
  const unsigned short* xbase[2];
  unsigned aoff[2][2], boff[2][2];           // [cc][buf] LDS byte offsets
  #pragma unroll
  for (int cc = 0; cc < 2; ++cc) {
    const int c  = tid + cc * 256;           // chunk id 0..511
    const int m  = c >> 2;                   // row within tile
    const int ko = (c & 3) * 8;              // k element offset
    agbase[cc] = aggw + (((size_t)(b * COUT + by * MT + m)) * 9) * CIN + ko;
    const int ng = bx * NT + m;              // global pixel for B rows
    const int y  = ng >> 6, xc = ng & 63;
    xbase[cc]  = xpad + (((size_t)(b * PADW + y)) * PADW + xc) * CIN + ko;
    aoff[cc][0] = (unsigned)(uintptr_t)&Asm[0][m * KT + ko];
    aoff[cc][1] = (unsigned)(uintptr_t)&Asm[1][m * KT + ko];
    boff[cc][0] = (unsigned)(uintptr_t)&Bsm[0][m * KT + ko];
    boff[cc][1] = (unsigned)(uintptr_t)&Bsm[1][m * KT + ko];
  }

  auto issue = [&](int step, int buf) {
    const int rs = step >> 3;                // 0..8
    const int cin0 = (step & 7) * KT;
    const int r = rs / 3, s = rs - r * 3;
    const size_t aE = (size_t)rs * CIN + cin0;            // uniform elem offsets
    const size_t bE = ((size_t)r * PADW + s) * CIN + cin0;
    #pragma unroll
    for (int cc = 0; cc < 2; ++cc) async_b128_to_lds(aoff[cc][buf], agbase[cc] + aE);
    #pragma unroll
    for (int cc = 0; cc < 2; ++cc) async_b128_to_lds(boff[cc][buf], xbase[cc] + bE);
  };

  v8f acc[2][4] = {};                        // 64 VGPRs of f32 accumulators
  const int lhalf = lane >> 4;               // 0: lanes 0-15, 1: lanes 16-31
  const int l15   = lane & 15;

  issue(0, 0);
  for (int step = 0; step < NSTEP; ++step) {
    const int buf = step & 1;
    wait_async0();                           // this step's tiles landed in LDS
    __syncthreads();                         // all waves' tiles visible
    if (step + 1 < NSTEP) issue(step + 1, buf ^ 1);  // prefetch next K-step

    // ---- fragments per documented 16-bit A/B VGPR striping ----
    union { v16bf v; uint4 u[2]; } afrag[2], bfrag[4];
    #pragma unroll
    for (int i = 0; i < 2; ++i) {
      const int m = wm * 32 + i * 16 + l15;
      const uint4* p = (const uint4*)(&Asm[buf][m * KT]);
      afrag[i].u[0] = p[lhalf];              // K 0-7   | 8-15
      afrag[i].u[1] = p[lhalf + 2];          // K 16-23 | 24-31
    }
    #pragma unroll
    for (int j = 0; j < 4; ++j) {
      const int n = wn * 64 + j * 16 + l15;
      const uint4* p = (const uint4*)(&Bsm[buf][n * KT]);
      bfrag[j].u[0] = p[lhalf];              // K 0-7   | 8-15
      bfrag[j].u[1] = p[lhalf + 2];          // K 16-23 | 24-31
    }
    #pragma unroll
    for (int i = 0; i < 2; ++i)
      #pragma unroll
      for (int j = 0; j < 4; ++j)
        acc[i][j] = __builtin_amdgcn_wmma_f32_16x16x32_bf16(
            false, afrag[i].v, false, bfrag[j].v,
            (short)0, acc[i][j], false, false);
    // no trailing barrier: buffer 'buf' is only refilled at step+2's issue(),
    // which every wave reaches only after the step+1 barrier (post-compute).
  }

  // ---- epilogue: C/D layout — VGPR v: lanes0-15 M=v, lanes16-31 M=v+8 ----
  #pragma unroll
  for (int i = 0; i < 2; ++i) {
    const int obase = by * MT + wm * 32 + i * 16 + lhalf * 8;
    #pragma unroll
    for (int j = 0; j < 4; ++j) {
      const int n = bx * NT + wn * 64 + j * 16 + l15;
      #pragma unroll
      for (int v = 0; v < 8; ++v)
        out[((size_t)(b * COUT + obase + v)) * NPIX + n] = acc[i][j][v];
    }
  }
}

// ---------------------------------------------------------------------------
extern "C" void kernel_launch(void* const* d_in, const int* in_sizes, int n_in,
                              void* d_out, int out_size, void* d_ws, size_t ws_size,
                              hipStream_t stream) {
  const float* x      = (const float*)d_in[0];
  const float* w1     = (const float*)d_in[1];
  const float* gamma  = (const float*)d_in[2];
  const float* beta   = (const float*)d_in[3];
  const float* mean   = (const float*)d_in[4];
  const float* var    = (const float*)d_in[5];
  const float* w2     = (const float*)d_in[6];
  const float* weight = (const float*)d_in[7];
  float* out = (float*)d_out;

  char* ws = (char*)d_ws;
  float* att    = (float*)ws;                                   // 64 f32
  float* pooled = (float*)(ws + 1024);                          // 16 KB
  const size_t xpad_bytes = (size_t)B_ * PADW * PADW * CIN * 2; // ~34 MB
  unsigned short* xpad = (unsigned short*)(ws + 32768);
  unsigned short* aggw = (unsigned short*)(ws + 32768 + xpad_bytes); // ~19 MB

  hipMemsetAsync(xpad, 0, xpad_bytes, stream);                  // zero pad halo

  pool_kernel<<<B_ * CIN, 256, 0, stream>>>(x, pooled);
  att_kernel <<<1, 256, 0, stream>>>(pooled, w1, gamma, beta, mean, var, w2, att);
  xpad_kernel<<<dim3(HW, CIN, B_), 64, 0, stream>>>(x, xpad);
  aggw_kernel<<<dim3(COUT, B_), 256, 0, stream>>>(weight, att, aggw);
  conv_kernel<<<dim3(NPIX / NT, COUT / MT, B_), 256, 0, stream>>>(xpad, aggw, out);
}
